// UpBlock_77017353551948
// MI455X (gfx1250) — matrix operand order repaired and checked by reading, not compiled, not run on hardware
//
#include <hip/hip_runtime.h>
#include <hip/hip_bf16.h>

// ---------------------------------------------------------------------------
// Types for CDNA5 WMMA (wave32): A/B = 16 halves/lane, C/D = 8 floats/lane
// ---------------------------------------------------------------------------
typedef __attribute__((ext_vector_type(16))) _Float16 v16h;
typedef __attribute__((ext_vector_type(8)))  _Float16 v8h;
typedef __attribute__((ext_vector_type(4)))  _Float16 v4h;
typedef __attribute__((ext_vector_type(8)))  float    v8f;
typedef __attribute__((ext_vector_type(4)))  float    v4f;

// Problem constants (from reference)
#define BB   8
#define SS   2048
#define DD   512
#define HH   2
#define DK   512
#define HD   1024          // H*DK
#define MM   (BB*SS)       // 16384 rows
#define OUTD 512

__device__ __forceinline__ v8f wmma16(v16h a, v16h b, v8f c) {
  return __builtin_amdgcn_wmma_f32_16x16x32_f16(false, a, false, b, (short)0, c,
                                                false, false);
}

__device__ __forceinline__ v16h cat8(v8h lo, v8h hi) {
  return __builtin_shufflevector(lo, hi, 0, 1, 2, 3, 4, 5, 6, 7,
                                 8, 9, 10, 11, 12, 13, 14, 15);
}

// A-fragment 16x32 (MxK).  Lanes 0-15: row=lane, K 0-15;
// lanes 16-31: row=lane-16, K 16-31.  ld in halves (multiple of 8).
__device__ __forceinline__ v16h load_a(const _Float16* base, int ld) {
  int lane = threadIdx.x & 31;
  int m  = lane & 15;
  int kb = (lane & 16) ? 16 : 0;
  const _Float16* p = base + (size_t)m * ld + kb;
  return cat8(*(const v8h*)(p), *(const v8h*)(p + 8));
}

// B-fragment 32x16 (KxN) from row-major [N,K] storage (A @ W^T pattern).
// Lane = column n; lanes 0-15 K=0-15, lanes 16-31 K=16-31.
__device__ __forceinline__ v16h load_b_nk(const _Float16* base, int ld) {
  int lane = threadIdx.x & 31;
  int n  = lane & 15;
  int kb = (lane & 16) ? 16 : 0;
  const _Float16* p = base + (size_t)n * ld + kb;
  return cat8(*(const v8h*)(p), *(const v8h*)(p + 8));
}

// 32B-aligned variant for global sources with ld % 16 == 0.
__device__ __forceinline__ v16h load_b_nk_g(const _Float16* base, int ld) {
  int lane = threadIdx.x & 31;
  int n  = lane & 15;
  int kb = (lane & 16) ? 16 : 0;
  return *(const v16h*)(base + (size_t)n * ld + kb);
}

// ---------------------------------------------------------------------------
// Tensor Data Mover (CDNA5): async 2D tile load global -> LDS, padded pitch.
// Guarded: falls back to manual LDS staging when the builtin is absent.
// ---------------------------------------------------------------------------
#if __has_builtin(__builtin_amdgcn_tensor_load_to_lds)
#define HAVE_TDM 1
typedef __attribute__((ext_vector_type(4))) unsigned int v4u_t;
typedef __attribute__((ext_vector_type(8))) int          v8i_t;
typedef __attribute__((ext_vector_type(4))) int          v4i_t;

// rows x row_dwords tile (4-byte elements), source row pitch row_stride_dwords.
// LDS pitch = (16 + 4) dwords per row via pad_interval=16dw / pad_amount=4dw.
__device__ __forceinline__ void tdm_load_2d(const void* gaddr, void* lds,
                                            int rows, int row_dwords,
                                            int row_stride_dwords) {
  unsigned long long ga = (unsigned long long)(uintptr_t)gaddr;
  unsigned lds_off = (unsigned)(uintptr_t)lds;   // low 32 bits = LDS byte addr
  v4u_t g0;
  g0[0] = 1u;                                        // count=1 (valid user D#)
  g0[1] = lds_off;                                   // lds_addr
  g0[2] = (unsigned)(ga & 0xffffffffu);              // global_addr[31:0]
  g0[3] = (unsigned)((ga >> 32) & 0x01ffffffu) |     // global_addr[56:32]
          (2u << 30);                                // type = 2 (image)
  unsigned td0 = (unsigned)row_stride_dwords;        // tensor_dim0 (no clip)
  unsigned td1 = (unsigned)rows;                     // tensor_dim1
  v8i_t g1;
  g1[0] = (int)((2u << 16) |                         // data_size = 4B
                (1u << 20) |                         // pad_enable
                (3u << 22) |                         // pad_interval: 16 dwords
                (3u << 25));                         // pad_amount: 4 dwords
  g1[1] = (int)((td0 & 0xffffu) << 16);              // tensor_dim0[15:0]
  g1[2] = (int)(((td0 >> 16) & 0xffffu) |
                ((td1 & 0xffffu) << 16));            // dim0[31:16] | dim1[15:0]
  g1[3] = (int)(((td1 >> 16) & 0xffffu) |
                (((unsigned)row_dwords & 0xffffu) << 16)); // | tile_dim0
  g1[4] = (int)((unsigned)rows & 0xffffu);           // tile_dim1 (tile_dim2=0)
  g1[5] = (int)td0;                                  // tensor_dim0_stride lo
  g1[6] = 0;
  g1[7] = 0;
  v4i_t z4 = {0, 0, 0, 0};
#if defined(__clang_major__) && (__clang_major__ >= 23)
  v8i_t z8 = {0, 0, 0, 0, 0, 0, 0, 0};
  __builtin_amdgcn_tensor_load_to_lds(g0, g1, z4, z4, z8, 0);
#else
  __builtin_amdgcn_tensor_load_to_lds(g0, g1, z4, z4, 0);
#endif
}
#else
#define HAVE_TDM 0
#endif

// ---------------------------------------------------------------------------
// f32 -> f16 convert (4-wide)
// ---------------------------------------------------------------------------
__global__ void f2h_kernel(const float* __restrict__ in,
                           _Float16* __restrict__ out, int n4) {
  int i = blockIdx.x * 256 + threadIdx.x;
  int stride = gridDim.x * 256;
  for (; i < n4; i += stride) {
    v4f v = *(const v4f*)(in + (size_t)i * 4);
    v4h h;
    h[0] = (_Float16)v[0]; h[1] = (_Float16)v[1];
    h[2] = (_Float16)v[2]; h[3] = (_Float16)v[3];
    *(v4h*)(out + (size_t)i * 4) = h;
  }
}

// ---------------------------------------------------------------------------
// GEMM: C[M,N] = act(A[M,K] @ W[N,K]^T + bias)
// Block = 256 threads (8 waves); tile 128(M) x 64(N); wave = 32x32 (2x2 frags)
// Double-buffered LDS staging (TDM async tiles when available).
// ACT: 0 = none, 1 = SiLU
// ---------------------------------------------------------------------------
#define GLDA 40
#define GLDW 40

template <int ACT, typename OT>
__global__ __launch_bounds__(256, 1) void gemm_wmma(
    const _Float16* __restrict__ A, const _Float16* __restrict__ W,
    const float* __restrict__ bias, OT* __restrict__ C, int M, int N, int K) {
  __shared__ _Float16 As[2][128 * GLDA];
  __shared__ _Float16 Ws[2][64 * GLDW];

  int tid  = threadIdx.x;
  int wave = tid >> 5;
  int lane = tid & 31;
  int m0 = blockIdx.y * 128;
  int n0 = blockIdx.x * 64;
  int wm = (wave & 3) * 32;
  int wn = (wave >> 2) * 32;

  v8f acc[2][2] = {};
  const int NK = K >> 5;

#if HAVE_TDM
  auto tdm_issue = [&](int buf, int ki) {
    int k0 = ki * 32;
    tdm_load_2d(&A[(size_t)m0 * K + k0], (void*)&As[buf][0], 128, 16, K >> 1);
    tdm_load_2d(&W[(size_t)n0 * K + k0], (void*)&Ws[buf][0], 64, 16, K >> 1);
  };
#else
  auto stage = [&](int buf, int ki) {
    int k0 = ki * 32;
    for (int c = tid; c < 512; c += 256) {
      int r = c >> 2, off = (c & 3) * 8;
      *(v8h*)&As[buf][r * GLDA + off] =
          *(const v8h*)&A[(size_t)(m0 + r) * K + k0 + off];
    }
    int r = tid >> 2, off = (tid & 3) * 8;
    *(v8h*)&Ws[buf][r * GLDW + off] =
        *(const v8h*)&W[(size_t)(n0 + r) * K + k0 + off];
  };
#endif

#if HAVE_TDM
  if (tid == 0) tdm_issue(0, 0);
  for (int ki = 0; ki < NK; ++ki) {
    int cur = ki & 1, nxt = cur ^ 1;
    if (tid == 0) __builtin_amdgcn_s_wait_tensorcnt(0);
    __syncthreads();                       // buf[cur] ready for everyone
    if (tid == 0 && ki + 1 < NK) tdm_issue(nxt, ki + 1);  // overlap with wmma
#else
  stage(0, 0);
  for (int ki = 0; ki < NK; ++ki) {
    int cur = ki & 1, nxt = cur ^ 1;
    __syncthreads();                       // buf[cur] ready
    if (ki + 1 < NK) stage(nxt, ki + 1);   // prefetch next tile during compute
#endif
    const _Float16* Ab = As[cur];
    const _Float16* Wb = Ws[cur];
    v16h bfr0 = load_b_nk(&Wb[(wn +  0) * GLDW], GLDW);
    v16h bfr1 = load_b_nk(&Wb[(wn + 16) * GLDW], GLDW);
#pragma unroll
    for (int i = 0; i < 2; ++i) {
      v16h a = load_a(&Ab[(wm + i * 16) * GLDA], GLDA);
      acc[i][0] = wmma16(a, bfr0, acc[i][0]);
      acc[i][1] = wmma16(a, bfr1, acc[i][1]);
    }
  }

  int cn = lane & 15;
  int rb = (lane & 16) ? 8 : 0;
#pragma unroll
  for (int i = 0; i < 2; ++i)
#pragma unroll
    for (int j = 0; j < 2; ++j) {
      int col = n0 + wn + j * 16 + cn;
      float bv = bias[col];
#pragma unroll
      for (int r = 0; r < 8; ++r) {
        float v = acc[i][j][r] + bv;
        if (ACT == 1) v = v / (1.0f + __expf(-v));   // SiLU
        int row = m0 + wm + i * 16 + rb + r;
        C[(size_t)row * N + col] = (OT)v;
      }
    }
}

// ---------------------------------------------------------------------------
// V transpose: v[b,s,h,d] -> vT[(b,h), d, s]   (tiled 32x32 through LDS)
// ---------------------------------------------------------------------------
__global__ void transpose_v(const _Float16* __restrict__ V,
                            _Float16* __restrict__ VT) {
  __shared__ _Float16 tile[32][33];
  int bh = blockIdx.z;
  int b = bh >> 1, h = bh & 1;
  int key0 = blockIdx.x * 32;
  int d0   = blockIdx.y * 32;
  int tx = threadIdx.x & 31, ty = threadIdx.x >> 5;
  const _Float16* src = V + ((size_t)b * SS) * HD + h * DK;
#pragma unroll
  for (int j = 0; j < 4; ++j)
    tile[ty + j * 8][tx] = src[(size_t)(key0 + ty + j * 8) * HD + d0 + tx];
  __syncthreads();
  _Float16* dst = VT + ((size_t)bh * DK) * SS;
#pragma unroll
  for (int j = 0; j < 4; ++j)
    dst[(size_t)(d0 + ty + j * 8) * SS + key0 + tx] = tile[tx][ty + j * 8];
}

// ---------------------------------------------------------------------------
// Flash attention (streaming keys, online softmax; scores never hit memory)
// Block = 256 threads; QT=64 query rows; KT=64 keys per step.
// S-tile 64x64 = 16 frags: wave w -> rows (w&3)*16, cols (w>>2)*32 (2 frags
// sharing one A fragment -> 3 loads : 2 wmma, software-pipelined).
// O accum: wave w owns output dims [w*64, w*64+64): 4x4 f32 fragments.
// ---------------------------------------------------------------------------
#define QT 64
#define KT 64
#define SLD 68    // 64 + 4 pad (floats)
#define PLD 72    // 64 + 8 pad (halves)

__global__ __launch_bounds__(256, 1) void flash_attn(
    const _Float16* __restrict__ Q, const _Float16* __restrict__ Kb,
    const _Float16* __restrict__ VT, _Float16* __restrict__ O) {
  __shared__ float    Ssm[QT * SLD];
  __shared__ _Float16 Ps[QT * PLD];
  __shared__ float    mrow[QT], lrow[QT], frow[QT];

  int tid  = threadIdx.x;
  int lane = tid & 31;
  int wave = tid >> 5;
  int q0 = blockIdx.x * QT;
  int bh = blockIdx.y;
  int b  = bh >> 1;
  int h  = bh & 1;

  const _Float16* Qbase  = Q  + ((size_t)b * SS + q0) * HD + h * DK;
  const _Float16* Kbase  = Kb + ((size_t)b * SS) * HD + h * DK;
  const _Float16* VTbase = VT + ((size_t)bh * DK) * SS;

  int smi = (wave & 3) * 16;     // S rows for this wave
  int sn0 = (wave >> 2) * 32;    // S cols (two 16-wide frags)
  int wc0 = wave * 64;           // O-dim slice owned by this wave

  v8f oacc[4][4] = {};
  if (tid < QT) { mrow[tid] = -1e30f; lrow[tid] = 0.0f; }

  const float scale = 0.044194173824159216f;  // 1/sqrt(512)
  int rb = (lane & 16) ? 8 : 0;
  int cn = lane & 15;

  for (int kt = 0; kt < SS; kt += KT) {
    // S frags = Q_tile @ K_tile^T  (16 steps over K=512, A+B pipelined)
    v8f sfr0 = {}, sfr1 = {};
    {
      const _Float16* qp  = Qbase + (size_t)smi * HD;
      const _Float16* kp0 = Kbase + (size_t)(kt + sn0) * HD;
      const _Float16* kp1 = kp0 + (size_t)16 * HD;
      v16h a  = load_a(qp, HD);
      v16h b0 = load_b_nk_g(kp0, HD);
      v16h b1 = load_b_nk_g(kp1, HD);
#pragma unroll
      for (int kk = 0; kk < 16; ++kk) {
        v16h an = a, b0n = b0, b1n = b1;
        if (kk < 15) {
          an  = load_a(qp + (kk + 1) * 32, HD);
          b0n = load_b_nk_g(kp0 + (kk + 1) * 32, HD);
          b1n = load_b_nk_g(kp1 + (kk + 1) * 32, HD);
        }
        sfr0 = wmma16(a, b0, sfr0);
        sfr1 = wmma16(a, b1, sfr1);
        a = an; b0 = b0n; b1 = b1n;
      }
    }
#pragma unroll
    for (int r = 0; r < 8; ++r) {
      Ssm[(smi + rb + r) * SLD + sn0 + cn]      = sfr0[r];
      Ssm[(smi + rb + r) * SLD + sn0 + 16 + cn] = sfr1[r];
    }

    __syncthreads();  // (B) S tile ready

    // online softmax: one thread per query row (two-pass over LDS)
    if (tid < QT) {
      float m_old = mrow[tid], l_old = lrow[tid];
      float mt = -1e30f;
#pragma unroll
      for (int j = 0; j < KT; ++j) mt = fmaxf(mt, Ssm[tid * SLD + j]);
      float mn  = fmaxf(m_old, mt * scale);
      float fac = __expf(m_old - mn);
      float lsum = 0.0f;
#pragma unroll
      for (int j = 0; j < KT; ++j) {
        float p = __expf(Ssm[tid * SLD + j] * scale - mn);
        lsum += p;
        Ps[tid * PLD + j] = (_Float16)p;
      }
      mrow[tid] = mn;
      lrow[tid] = l_old * fac + lsum;
      frow[tid] = fac;
    }
    __syncthreads();  // (C) P + rescale factors ready

    // rescale running O accumulators
#pragma unroll
    for (int mi = 0; mi < 4; ++mi) {
      float f[8];
#pragma unroll
      for (int r = 0; r < 8; ++r) f[r] = frow[mi * 16 + rb + r];
#pragma unroll
      for (int nj = 0; nj < 4; ++nj)
#pragma unroll
        for (int r = 0; r < 8; ++r) oacc[mi][nj][r] *= f[r];
    }

    // O += P(64x64) @ V(64x64-slice): 2 k-steps, B pipelined
#pragma unroll
    for (int kk2 = 0; kk2 < 2; ++kk2) {
      v16h pa[4];
#pragma unroll
      for (int mi = 0; mi < 4; ++mi)
        pa[mi] = load_a(&Ps[(mi * 16) * PLD + kk2 * 32], PLD);
      v16h bvv = load_b_nk_g(VTbase + (size_t)wc0 * SS + kt + kk2 * 32, SS);
#pragma unroll
      for (int nj = 0; nj < 4; ++nj) {
        v16h b2 = bvv;
        if (nj < 3)
          b2 = load_b_nk_g(VTbase + (size_t)(wc0 + (nj + 1) * 16) * SS +
                               kt + kk2 * 32, SS);
#pragma unroll
        for (int mi = 0; mi < 4; ++mi)
          oacc[mi][nj] = wmma16(pa[mi], bvv, oacc[mi][nj]);
        bvv = b2;
      }
    }
  }

  __syncthreads();
  // finalize: divide by l, store f16
#pragma unroll
  for (int mi = 0; mi < 4; ++mi) {
    float inv[8];
#pragma unroll
    for (int r = 0; r < 8; ++r) inv[r] = 1.0f / lrow[mi * 16 + rb + r];
#pragma unroll
    for (int nj = 0; nj < 4; ++nj)
#pragma unroll
      for (int r = 0; r < 8; ++r) {
        int row = q0 + mi * 16 + rb + r;
        int col = h * DK + wc0 + nj * 16 + cn;
        O[((size_t)b * SS + row) * HD + col] =
            (_Float16)(oacc[mi][nj][r] * inv[r]);
      }
  }
}

// ---------------------------------------------------------------------------
// LayerNorm over 512 + residual add (d_out already holds skip@Wp^T+bp)
// ---------------------------------------------------------------------------
__global__ void ln_residual(const float* __restrict__ H,
                            const float* __restrict__ gamma,
                            const float* __restrict__ beta,
                            float* __restrict__ Out) {
  __shared__ float sS[8], sQ[8], bc[2];
  int row = blockIdx.x;
  int tid = threadIdx.x;
  int lane = tid & 31, wave = tid >> 5;
  const float* h = H + (size_t)row * OUTD;
  float v0 = h[tid], v1 = h[tid + 256];
  float s  = v0 + v1;
  float sq = v0 * v0 + v1 * v1;
#pragma unroll
  for (int off = 16; off > 0; off >>= 1) {
    s  += __shfl_down(s,  off, 32);
    sq += __shfl_down(sq, off, 32);
  }
  if (lane == 0) { sS[wave] = s; sQ[wave] = sq; }
  __syncthreads();
  if (tid == 0) {
    float a = 0.0f, q = 0.0f;
#pragma unroll
    for (int i = 0; i < 8; ++i) { a += sS[i]; q += sQ[i]; }
    float mu  = a * (1.0f / OUTD);
    float var = q * (1.0f / OUTD) - mu * mu;
    bc[0] = mu;
    bc[1] = rsqrtf(var + 1e-5f);
  }
  __syncthreads();
  float mu = bc[0], inv = bc[1];
  float* o = Out + (size_t)row * OUTD;
  o[tid]       = (v0 - mu) * inv * gamma[tid]       + beta[tid]       + o[tid];
  o[tid + 256] = (v1 - mu) * inv * gamma[tid + 256] + beta[tid + 256] + o[tid + 256];
}

// ---------------------------------------------------------------------------
// Host-side orchestration
// ---------------------------------------------------------------------------
extern "C" void kernel_launch(void* const* d_in, const int* in_sizes, int n_in,
                              void* d_out, int out_size, void* d_ws, size_t ws_size,
                              hipStream_t stream) {
  const float* x     = (const float*)d_in[0];
  const float* skip  = (const float*)d_in[1];
  const float* Wq    = (const float*)d_in[2];
  const float* bq    = (const float*)d_in[3];
  const float* Wk    = (const float*)d_in[4];
  const float* bk    = (const float*)d_in[5];
  const float* Wv    = (const float*)d_in[6];
  const float* bv    = (const float*)d_in[7];
  const float* Wo    = (const float*)d_in[8];
  const float* bo    = (const float*)d_in[9];
  const float* Wl    = (const float*)d_in[10];
  const float* bl    = (const float*)d_in[11];
  const float* gamma = (const float*)d_in[12];
  const float* beta  = (const float*)d_in[13];
  const float* Wp    = (const float*)d_in[14];
  const float* bp    = (const float*)d_in[15];
  float* out = (float*)d_out;

  char* ws = (char*)d_ws;
  size_t off = 0;
  auto alloc = [&](size_t bytes) -> void* {
    void* p = ws + off;
    off = (off + bytes + 255) & ~(size_t)255;
    return p;
  };
  _Float16* xh    = (_Float16*)alloc((size_t)MM * DD * 2);
  _Float16* skiph = (_Float16*)alloc((size_t)MM * DD * 2);
  _Float16* Wqh   = (_Float16*)alloc((size_t)HD * DD * 2);
  _Float16* Wkh   = (_Float16*)alloc((size_t)HD * DD * 2);
  _Float16* Wvh   = (_Float16*)alloc((size_t)HD * DD * 2);
  _Float16* Woh   = (_Float16*)alloc((size_t)DD * HD * 2);
  _Float16* Wlh   = (_Float16*)alloc((size_t)OUTD * DD * 2);
  _Float16* Wph   = (_Float16*)alloc((size_t)OUTD * DD * 2);
  _Float16* qh    = (_Float16*)alloc((size_t)MM * HD * 2);
  _Float16* kh    = (_Float16*)alloc((size_t)MM * HD * 2);
  _Float16* vh    = (_Float16*)alloc((size_t)MM * HD * 2);
  _Float16* vth   = (_Float16*)alloc((size_t)MM * HD * 2);  // V transposed
  _Float16* oh    = (_Float16*)alloc((size_t)MM * HD * 2);
  _Float16* sh    = (_Float16*)alloc((size_t)MM * DD * 2);
  float*    hbuf  = (float*)alloc((size_t)MM * OUTD * 4);

  dim3 blk(256);

  f2h_kernel<<<dim3(512), blk, 0, stream>>>(x,    xh,    MM * DD / 4);
  f2h_kernel<<<dim3(512), blk, 0, stream>>>(skip, skiph, MM * DD / 4);
  f2h_kernel<<<dim3(128), blk, 0, stream>>>(Wq, Wqh, HD * DD / 4);
  f2h_kernel<<<dim3(128), blk, 0, stream>>>(Wk, Wkh, HD * DD / 4);
  f2h_kernel<<<dim3(128), blk, 0, stream>>>(Wv, Wvh, HD * DD / 4);
  f2h_kernel<<<dim3(128), blk, 0, stream>>>(Wo, Woh, DD * HD / 4);
  f2h_kernel<<<dim3(64),  blk, 0, stream>>>(Wl, Wlh, OUTD * DD / 4);
  f2h_kernel<<<dim3(64),  blk, 0, stream>>>(Wp, Wph, OUTD * DD / 4);

  // Q, K, V projections: [16384,512] @ [512,1024]
  dim3 gqkv(HD / 64, MM / 128);
  gemm_wmma<0, _Float16><<<gqkv, blk, 0, stream>>>(xh, Wqh, bq, qh, MM, HD, DD);
  gemm_wmma<0, _Float16><<<gqkv, blk, 0, stream>>>(xh, Wkh, bk, kh, MM, HD, DD);
  gemm_wmma<0, _Float16><<<gqkv, blk, 0, stream>>>(xh, Wvh, bv, vh, MM, HD, DD);

  // V -> V^T (per (b,h))
  transpose_v<<<dim3(SS / 32, DK / 32, BB * HH), blk, 0, stream>>>(vh, vth);

  // flash attention
  flash_attn<<<dim3(SS / QT, BB * HH), blk, 0, stream>>>(qh, kh, vth, oh);

  // o @ Wo^T + bo, fused SiLU -> sh (f16)
  dim3 go(DD / 64, MM / 128);
  gemm_wmma<1, _Float16><<<go, blk, 0, stream>>>(oh, Woh, bo, sh, MM, DD, HD);

  // silu(o) @ Wl^T + bl -> hbuf (f32)
  dim3 gl(OUTD / 64, MM / 128);
  gemm_wmma<0, float><<<gl, blk, 0, stream>>>(sh, Wlh, bl, hbuf, MM, OUTD, DD);

  // skip @ Wp^T + bp -> d_out (f32)
  gemm_wmma<0, float><<<gl, blk, 0, stream>>>(skiph, Wph, bp, out, MM, OUTD, DD);

  // LayerNorm(hbuf)*gamma+beta + d_out -> d_out
  ln_residual<<<dim3(MM), blk, 0, stream>>>(hbuf, gamma, beta, out);
}
